// BAGCN_77335181131827
// MI455X (gfx1250) — matrix operand order repaired and checked by reading, not compiled, not run on hardware
//
#include <hip/hip_runtime.h>
#include <hip/hip_bf16.h>
#include <stdint.h>

// BAGCN on MI455X (gfx1250, wave32).
//  - Dense GEMMs on v_wmma_f32_16x16x32_bf16 (fp32 accumulate).
//  - Weights pre-converted to bf16 once, staged per-K-slice into LDS via
//    GLOBAL_LOAD_ASYNC_TO_LDS_B128 (ASYNCcnt path) when available.
//  - SpMM via global_atomic_add_f32 scatter (x matrix is L2-resident: 153.6MB < 192MB L2).
//  - Row L2-normalize with wave32 shuffle reduction.
// Scratch layout in d_ws:
//   slot0: x (also SpMM accumulator), slot1/slot2: ping-pong b / (xw|b_next),
//   tail:  bf16 copies of node_ws and rel_ws (6 * 64K * 2B = 768KB).

typedef __attribute__((ext_vector_type(16))) __bf16 v16bf;
typedef __attribute__((ext_vector_type(8)))  float  v8f;
typedef __attribute__((ext_vector_type(4)))  int    v4i;

typedef __attribute__((address_space(1))) v4i gv4i;   // global int4
typedef __attribute__((address_space(3))) v4i lv4i;   // LDS int4

#define DD     256   // embedding dim
#define TILE_M 128   // rows per block (8 waves x 16 rows)
#define KSTEP  32    // WMMA K per step
#define NT     16    // 16 N-tiles of 16 = full 256 columns per wave
#define LWS    40    // padded LDS row stride in bf16 (80B) to spread banks

// ---------------------------------------------------------------------------
// f32 -> bf16 convert (weights), once per launch
// ---------------------------------------------------------------------------
__global__ __launch_bounds__(256)
void bagcn_cvt_bf16(const float* __restrict__ src, __bf16* __restrict__ dst, int n)
{
    const int i = blockIdx.x * blockDim.x + threadIdx.x;
    if (i < n) dst[i] = (__bf16)src[i];
}

// ---------------------------------------------------------------------------
// GEMM: out[m,n] = sum_k (A0[m,k] (+A1[m,k])) * W[n,k]   (i.e. A @ W^T)
// Wb is the bf16 copy of W (row-major [256,256]).
// Optionally outAcc[m,n] += accScale * out[m,n]  (fused behaviors update)
// ---------------------------------------------------------------------------
__global__ __launch_bounds__(256)
void bagcn_gemm_bf16(const float* __restrict__ A0,
                     const float* __restrict__ A1,
                     const __bf16* __restrict__ Wb,
                     float* __restrict__ outMain,
                     float* __restrict__ outAcc,
                     float accScale, int nRows, int addA1, int doAcc)
{
    __shared__ __bf16 lw[DD * LWS];     // 20KB: W k-slice, bf16

    const int tid   = threadIdx.x;
    const int lane  = tid & 31;
    const int waveI = tid >> 5;
    const int half  = lane >> 4;        // 0: lanes 0-15, 1: lanes 16-31
    const int l15   = lane & 15;
    const int mBase = blockIdx.x * TILE_M + waveI * 16;
    const int mRow  = mBase + l15;      // A-matrix row this lane feeds
    const bool mOk  = (mRow < nRows);

    v8f acc[NT];
#pragma unroll
    for (int t = 0; t < NT; ++t) acc[t] = (v8f){0.f,0.f,0.f,0.f,0.f,0.f,0.f,0.f};

    for (int kt = 0; kt < DD / KSTEP; ++kt) {
        __syncthreads();  // protect previous iteration's lw readers
        // ---- stage Wb[n=tid][kt*32 .. +31] -> LDS (one 64B row per thread)
        {
            const __bf16* wsrc = Wb + (size_t)tid * DD + kt * KSTEP;
            __bf16*       wdst = &lw[tid * LWS];
#if defined(__gfx1250__) && __has_builtin(__builtin_amdgcn_global_load_async_to_lds_b128)
#pragma unroll
            for (int q = 0; q < 4; ++q)
                __builtin_amdgcn_global_load_async_to_lds_b128(
                    (gv4i*)(wsrc + q * 8),   // global src (cast drops const)
                    (lv4i*)(wdst + q * 8),   // LDS dst
                    0, 0);
#if __has_builtin(__builtin_amdgcn_s_wait_asynccnt)
            __builtin_amdgcn_s_wait_asynccnt(0);
#endif
#else
#pragma unroll
            for (int j = 0; j < KSTEP; ++j) wdst[j] = wsrc[j];
#endif
        }
        __syncthreads();

        // ---- A fragment, ISA 16x32 bf16 layout:
        //  lane half=0: elems 0..7 -> K k0..k0+7,  elems 8..15 -> K k0+16..k0+23 (k0=kt*32)
        //  lane half=1: same with k0+8
        v16bf afrag;
        {
            const int k0 = kt * KSTEP + half * 8;
            float fa[16];
            if (mOk) {
                const float* a0 = A0 + (size_t)mRow * DD + k0;
#pragma unroll
                for (int j = 0; j < 8; ++j) fa[j]     = a0[j];
#pragma unroll
                for (int j = 0; j < 8; ++j) fa[8 + j] = a0[16 + j];
                if (addA1) {
                    const float* a1 = A1 + (size_t)mRow * DD + k0;
#pragma unroll
                    for (int j = 0; j < 8; ++j) fa[j]     += a1[j];
#pragma unroll
                    for (int j = 0; j < 8; ++j) fa[8 + j] += a1[16 + j];
                }
            } else {
#pragma unroll
                for (int j = 0; j < 16; ++j) fa[j] = 0.f;
            }
#pragma unroll
            for (int j = 0; j < 16; ++j) afrag[j] = (__bf16)fa[j];
        }

        // ---- 16 B fragments + WMMA. B lane l: col n = nt*16+l15,
        //      elems j=0..7 -> K = kt*32 + half*8 + j, j=8..15 -> K = kt*32 + 16 + half*8 + (j-8)
#pragma unroll
        for (int nt = 0; nt < NT; ++nt) {
            const __bf16* bp = &lw[(nt * 16 + l15) * LWS + half * 8];
            union { v16bf v; uint4 q[2]; } bu;
            bu.q[0] = *(const uint4*)(bp);
            bu.q[1] = *(const uint4*)(bp + 16);
            acc[nt] = __builtin_amdgcn_wmma_f32_16x16x32_bf16(
                false, afrag, false, bu.v, (short)0, acc[nt], false, false);
        }
    }

    // ---- epilogue. C layout: VGPR r -> row mBase + half*8 + r, col nt*16 + l15
#pragma unroll
    for (int nt = 0; nt < NT; ++nt) {
        const int col = nt * 16 + l15;
#pragma unroll
        for (int r = 0; r < 8; ++r) {
            const int row = mBase + half * 8 + r;
            if (row < nRows) {
                const size_t idx = (size_t)row * DD + col;
                const float vv = acc[nt][r];
                outMain[idx] = vv;
                if (doAcc) outAcc[idx] += accScale * vv;
            }
        }
    }
}

// ---------------------------------------------------------------------------
// SpMM scatter: Y[rows[e]] += vals[e] * X[cols[e]]   (one wave per edge)
// ---------------------------------------------------------------------------
__global__ __launch_bounds__(256)
void bagcn_spmm(const int* __restrict__ rows, const int* __restrict__ cols,
                const float* __restrict__ vals,
                const float* __restrict__ X, float* __restrict__ Y, int nnz)
{
    const int wave = blockIdx.x * (blockDim.x >> 5) + (threadIdx.x >> 5);
    const int lane = threadIdx.x & 31;
    if (wave >= nnz) return;

    const int   r = rows[wave];
    const int   c = cols[wave];
    const float v = vals[wave];

    const float4* xs = (const float4*)(X + (size_t)c * DD);
    float*        yr = Y + (size_t)r * DD;

#pragma unroll
    for (int i = 0; i < 2; ++i) {
        const int e4 = lane + i * 32;          // float4 index within the row
        const float4 x4 = xs[e4];
        float* y = yr + e4 * 4;
        __hip_atomic_fetch_add(y + 0, v * x4.x, __ATOMIC_RELAXED, __HIP_MEMORY_SCOPE_AGENT);
        __hip_atomic_fetch_add(y + 1, v * x4.y, __ATOMIC_RELAXED, __HIP_MEMORY_SCOPE_AGENT);
        __hip_atomic_fetch_add(y + 2, v * x4.z, __ATOMIC_RELAXED, __HIP_MEMORY_SCOPE_AGENT);
        __hip_atomic_fetch_add(y + 3, v * x4.w, __ATOMIC_RELAXED, __HIP_MEMORY_SCOPE_AGENT);
    }
}

// ---------------------------------------------------------------------------
// Row-wise L2 normalize in place + Res += scale * normalized  (one wave / row)
// ---------------------------------------------------------------------------
__global__ __launch_bounds__(256)
void bagcn_norm(float* __restrict__ X, float* __restrict__ Res,
                float scale, int nRows)
{
    const int row  = blockIdx.x * 8 + (threadIdx.x >> 5);
    const int lane = threadIdx.x & 31;
    if (row >= nRows) return;

    float4* xr = (float4*)(X + (size_t)row * DD);
    float4  x0 = xr[lane];
    float4  x1 = xr[lane + 32];

    float ss = x0.x*x0.x + x0.y*x0.y + x0.z*x0.z + x0.w*x0.w
             + x1.x*x1.x + x1.y*x1.y + x1.z*x1.z + x1.w*x1.w;
#pragma unroll
    for (int off = 16; off > 0; off >>= 1) ss += __shfl_xor(ss, off, 32);

    const float inv = 1.0f / fmaxf(sqrtf(ss), 1e-12f);

    x0.x *= inv; x0.y *= inv; x0.z *= inv; x0.w *= inv;
    x1.x *= inv; x1.y *= inv; x1.z *= inv; x1.w *= inv;
    xr[lane]      = x0;
    xr[lane + 32] = x1;

    float4* rr = (float4*)(Res + (size_t)row * DD);
    float4 r0 = rr[lane], r1 = rr[lane + 32];
    r0.x += scale * x0.x; r0.y += scale * x0.y; r0.z += scale * x0.z; r0.w += scale * x0.w;
    r1.x += scale * x1.x; r1.y += scale * x1.y; r1.z += scale * x1.z; r1.w += scale * x1.w;
    rr[lane]      = r0;
    rr[lane + 32] = r1;
}

// ---------------------------------------------------------------------------
__global__ __launch_bounds__(256)
void bagcn_zero(float4* __restrict__ p, size_t n4)
{
    const size_t i = (size_t)blockIdx.x * blockDim.x + threadIdx.x;
    if (i < n4) p[i] = make_float4(0.f, 0.f, 0.f, 0.f);
}

// ---------------------------------------------------------------------------
extern "C" void kernel_launch(void* const* d_in, const int* in_sizes, int n_in,
                              void* d_out, int out_size, void* d_ws, size_t ws_size,
                              hipStream_t stream)
{
    const float* in_embs  = (const float*)d_in[0];   // [N, D]
    const float* beh_embs = (const float*)d_in[1];   // [N, D]
    const float* node_ws  = (const float*)d_in[2];   // [L, D, D]
    const float* rel_ws   = (const float*)d_in[3];   // [L, D, D]
    const float* vals     = (const float*)d_in[4];   // [nnz]
    const int*   rows     = (const int*)d_in[5];
    const int*   cols     = (const int*)d_in[6];

    const int N    = in_sizes[0] / DD;   // 150000
    const int nnz  = in_sizes[4];
    const size_t slot = (size_t)N * DD;

    float*  x     = (float*)d_ws;                 // slot0: x / SpMM accumulator
    float*  bbuf0 = (float*)d_ws + slot;          // slot1
    float*  bbuf1 = (float*)d_ws + 2 * slot;      // slot2
    __bf16* wbf   = (__bf16*)((float*)d_ws + 3 * slot);  // bf16 weights (768KB)

    float* result    = (float*)d_out;            // [N, D]
    float* behaviors = result + slot;            // [N, D]

    // init running state
    (void)hipMemcpyAsync(x,         in_embs,  slot * sizeof(float), hipMemcpyDeviceToDevice, stream);
    (void)hipMemcpyAsync(result,    in_embs,  slot * sizeof(float), hipMemcpyDeviceToDevice, stream);
    (void)hipMemcpyAsync(bbuf0,     beh_embs, slot * sizeof(float), hipMemcpyDeviceToDevice, stream);
    (void)hipMemcpyAsync(behaviors, beh_embs, slot * sizeof(float), hipMemcpyDeviceToDevice, stream);

    // bf16 weight copies (node then rel), once
    const int wElems = in_sizes[2];              // L*D*D
    bagcn_cvt_bf16<<<(wElems + 255) / 256, 256, 0, stream>>>(node_ws, wbf, wElems);
    bagcn_cvt_bf16<<<(wElems + 255) / 256, 256, 0, stream>>>(rel_ws, wbf + wElems, wElems);

    const int mBlocks  = (N + TILE_M - 1) / TILE_M;
    const int spBlocks = (nnz + 7) / 8;            // 8 waves/block, 1 edge/wave
    const int nmBlocks = (N + 7) / 8;
    const size_t n4    = slot / 4;
    const int  zBlocks = (int)((n4 + 255) / 256);

    float* bcur   = bbuf0;
    float* bother = bbuf1;   // holds xw, then b_next

    for (int i = 0; i < 3; ++i) {
        const float layerScale = 1.0f / (float)(i + 1);

        // xw = (x + b) @ node_ws[i]^T
        bagcn_gemm_bf16<<<mBlocks, 256, 0, stream>>>(
            x, bcur, wbf + (size_t)i * DD * DD,
            bother, (float*)nullptr, 0.f, N, /*addA1=*/1, /*doAcc=*/0);

        // x <- 0 ; x[rows] += vals * xw[cols]
        bagcn_zero<<<zBlocks, 256, 0, stream>>>((float4*)x, n4);
        bagcn_spmm<<<spBlocks, 256, 0, stream>>>(rows, cols, vals, bother, x, nnz);

        // x <- normalize(x); result += x / (i+1)
        bagcn_norm<<<nmBlocks, 256, 0, stream>>>(x, result, layerScale, N);

        // b_next = b @ rel_ws[i]^T ; behaviors += b_next / (i+1)
        bagcn_gemm_bf16<<<mBlocks, 256, 0, stream>>>(
            bcur, (const float*)nullptr, wbf + (size_t)wElems + (size_t)i * DD * DD,
            bother, behaviors, layerScale, N, /*addA1=*/0, /*doAcc=*/1);

        // ping-pong b buffers
        float* t = bcur; bcur = bother; bother = t;
    }
}
